// PowerSTGAT_1228360646950
// MI455X (gfx1250) — compile-verified, matrix operand order checked by reading, not run on hardware
//
#include <hip/hip_runtime.h>
#include <hip/hip_bf16.h>

// ---------------------------------------------------------------------------
// PowerSTGAT on gfx1250 (MI455X): LSTM (WMMA f16 + TDM staging) -> 3x GATConv.
// Wave32; 16-node tiles; v_wmma_f32_16x16x32_f16 for all dense GEMMs.
// ---------------------------------------------------------------------------

typedef __attribute__((ext_vector_type(16))) _Float16 v16h;
typedef __attribute__((ext_vector_type(8)))  _Float16 v8h;
typedef __attribute__((ext_vector_type(8)))  float    v8f;

#define HD __device__ __forceinline__

// Hardware-native transcendental path: v_exp_f32 / v_rcp_f32 / v_rsq_f32.
HD float fexp(float x) { return __builtin_amdgcn_exp2f(x * 1.4426950408889634f); }
HD float frcp(float x) { return __builtin_amdgcn_rcpf(x); }
// sigmoid(x) = rcp(1 + e^-x); saturates correctly (exp->inf => 0, exp->0 => 1)
HD float sigf(float x) { return frcp(1.0f + fexp(-x)); }
// tanh(x) = 1 - 2*rcp(e^{2x} + 1); exp overflow -> inf -> rcp -> 0 -> 1.
HD float tanh_fast(float x) { return 1.0f - 2.0f * frcp(fexp(2.0f * x) + 1.0f); }

// Order-preserving float->uint encoding so integer atomicMax == float max.
HD unsigned encf(float f) {
    int i = __float_as_int(f);
    return (unsigned)(i ^ ((i >> 31) | 0x80000000));
}
HD float decf(unsigned u) {
    int i = (u & 0x80000000u) ? (int)(u ^ 0x80000000u) : ~(int)u;
    return __int_as_float(i);
}
#define ENC_NEG_INF 0x007FFFFFu  // encf(-inf)

// CDNA5 WMMA: D = A(16x32 f16) * B(32x16 f16) + C(16x16 f32)
HD v8f wmma_f16(v16h a, v16h b, v8f c) {
    return __builtin_amdgcn_wmma_f32_16x16x32_f16(
        /*neg_a=*/false, a, /*neg_b=*/false, b,
        /*c_mod=*/(short)0, c, /*reuse_a=*/false, /*reuse_b=*/false);
}

// ISA 16-bit A/B layout: per-lane runs of 8 contiguous K halves.
//   lanes 0-15 : K = [0..7],[16..23] (+32*kb);  lanes 16-31: K = [8..15],[24..31]
HD v16h load_tile16(const _Float16* base, int row, int lane, int kb) {
    int k0 = ((lane >> 4) << 3) + (kb << 5);
    v8h lo = *(const v8h*)(base + row * 64 + k0);
    v8h hi = *(const v8h*)(base + row * 64 + k0 + 16);
    return __builtin_shufflevector(lo, hi, 0, 1, 2, 3, 4, 5, 6, 7,
                                   8, 9, 10, 11, 12, 13, 14, 15);
}

// ---------------------------------------------------------------------------
// Tensor Data Mover staging of W_hh (fp16, 32 KB) into LDS.
// D# per CDNA5 ISA: one 1-row tile of 16384 2-byte elems.
// ---------------------------------------------------------------------------
#if defined(__has_builtin)
#if __has_builtin(__builtin_amdgcn_tensor_load_to_lds) && \
    __has_builtin(__builtin_amdgcn_s_wait_tensorcnt)
#define USE_TDM 1
#endif
#endif

#ifdef USE_TDM
typedef unsigned u32x4 __attribute__((ext_vector_type(4)));
typedef int      i32x4 __attribute__((ext_vector_type(4)));
typedef int      i32x8 __attribute__((ext_vector_type(8)));

HD void tdm_load_whh(const _Float16* gptr, _Float16* lptr) {
    const unsigned ELEMS = 16384u;  // 256*64 halves
    unsigned lds = (unsigned)(uintptr_t)lptr;  // LDS offset = flat addr[31:0]
    unsigned long long ga = (unsigned long long)(uintptr_t)gptr;
    u32x4 g0;
    g0[0] = 1u;                                   // count=1 (valid user D#)
    g0[1] = lds;                                  // lds_addr
    g0[2] = (unsigned)(ga & 0xFFFFFFFFu);         // global_addr[31:0]
    g0[3] = (unsigned)((ga >> 32) & 0x01FFFFFFu)  // global_addr[56:32]
            | (2u << 30);                         // type = 2 ("image")
    i32x8 g1;
    g1[0] = (int)(1u << 16);                      // data_size=1 (2 bytes)
    g1[1] = (int)((ELEMS & 0xFFFFu) << 16);       // tensor_dim0[15:0]
    g1[2] = (int)(((ELEMS >> 16) & 0xFFFFu)       // tensor_dim0[31:16]
            | (1u << 16));                        // tensor_dim1 = 1
    g1[3] = (int)((ELEMS & 0xFFFFu) << 16);       // tile_dim0 = 16384
    g1[4] = 1;                                    // tile_dim1 = 1
    g1[5] = (int)ELEMS;                           // tensor_dim0_stride lo32
    g1[6] = (int)((ELEMS & 0xFFFFu) << 16);       // stride0 hi | stride1[15:0]
    g1[7] = 0;                                    // stride1[47:16]
    i32x4 z4 = {0, 0, 0, 0};
#if defined(__clang_major__) && __clang_major__ >= 23
    i32x8 z8 = {0, 0, 0, 0, 0, 0, 0, 0};
    __builtin_amdgcn_tensor_load_to_lds(g0, g1, z4, z4, z8, 0);
#else
    __builtin_amdgcn_tensor_load_to_lds(g0, g1, z4, z4, 0);
#endif
    __builtin_amdgcn_s_wait_tensorcnt((short)0);
}
#endif

// ---------------------------------------------------------------------------
// Prep: W_hh fp32 -> fp16; packed {wih0, wih1, bias, 0} per gate column.
// ---------------------------------------------------------------------------
__global__ void lstm_prep_kernel(const float* __restrict__ Whh,
                                 const float* __restrict__ Wih,
                                 const float* __restrict__ bih,
                                 const float* __restrict__ bhh,
                                 _Float16* __restrict__ whh16,
                                 float4* __restrict__ wbB) {
    int idx = blockIdx.x * blockDim.x + threadIdx.x;
    if (idx < 256 * 64) whh16[idx] = (_Float16)Whh[idx];
    if (idx < 256) {
        float4 v;
        v.x = Wih[idx * 2 + 0];
        v.y = Wih[idx * 2 + 1];
        v.z = bih[idx] + bhh[idx];
        v.w = 0.f;
        wbB[idx] = v;
    }
}

// ---------------------------------------------------------------------------
// LSTM: 4 waves/block, each wave owns a 16-node tile; one shared W_hh copy.
// gates[16,256] = x_t@Wih^T + h@Whh^T + b.  h recurs through LDS as fp16.
// ---------------------------------------------------------------------------
#define LSTM_WAVES 4
__global__ __launch_bounds__(32 * LSTM_WAVES, 4) void lstm_kernel(
    const float* __restrict__ x,          // [N,24,2]
    const float4* __restrict__ wbB,       // [256] {wih0,wih1,bias,0}
    const _Float16* __restrict__ whh16,   // [256][64] (gate row, hidden k)
    _Float16* __restrict__ featH,         // [N,64] last hidden (fp16)
    int N) {
    __shared__ __align__(16) _Float16 sWhh[256 * 64];          // 32 KB shared
    __shared__ __align__(16) float4   sWb[256];                // 4 KB
    __shared__ __align__(16) float    sx[LSTM_WAVES][16 * 24 * 2];
    __shared__ __align__(16) _Float16 sh[LSTM_WAVES][16 * 64];
    const int tid = threadIdx.x;
    const int wave = tid >> 5;
    const int lane = tid & 31;
    const int tile = blockIdx.x * LSTM_WAVES + wave;
    const int NT = N / 16;

    // Stage shared W_hh: TDM async tile copy (wave 0) or cooperative loads.
#ifdef USE_TDM
    if (wave == 0) tdm_load_whh(whh16, sWhh);
#else
    {
        const uint4* src = (const uint4*)whh16;
        uint4* dst = (uint4*)sWhh;
        for (int i = tid; i < 2048; i += 32 * LSTM_WAVES) dst[i] = src[i];
    }
#endif
    for (int i = tid; i < 256; i += 32 * LSTM_WAVES) sWb[i] = wbB[i];
    if (tile < NT) {
        const float* src = x + (size_t)tile * 16 * 48;
        for (int i = lane; i < 768; i += 32) sx[wave][i] = src[i];
        for (int i = lane; i < 1024; i += 32) sh[wave][i] = (_Float16)0.f;
    }
    __syncthreads();
    if (tile >= NT) return;

    const int mhalf = (lane >> 4) << 3;  // C/D layout: node m = r + mhalf
    float cs[4][8];
#pragma unroll
    for (int j = 0; j < 4; ++j)
#pragma unroll
        for (int r = 0; r < 8; ++r) cs[j][r] = 0.f;

#pragma unroll 1  // recurrence: no cross-step pipelining, keep VGPRs <= 256
    for (int t = 0; t < 24; ++t) {
        // Compiler-level fence: keeps the (loop-invariant) W_hh B-tile loads
        // inside the loop as per-step ds_load_b128 instead of LICM-hoisting
        // them into capped VGPRs and spilling to scratch.  No HW effect.
        asm volatile("" ::: "memory");

        // A = h (16x64 fp16), shared by all 16 gate tiles this step.
        v16h hA0 = load_tile16(sh[wave], lane & 15, lane, 0);
        v16h hA1 = load_tile16(sh[wave], lane & 15, lane, 1);

        // x_t for the 8 nodes this lane's accumulator VGPRs cover.
        float xm0[8], xm1[8];
#pragma unroll
        for (int r = 0; r < 8; ++r) {
            int m = r + mhalf;
            float2 xp = *(const float2*)(&sx[wave][(m * 24 + t) * 2]);
            xm0[r] = xp.x;
            xm1[r] = xp.y;
        }

#pragma unroll
        for (int j = 0; j < 4; ++j) {  // hidden-unit chunk (16 wide)
            v8f acc[4];                // gate tiles: i, f, g, o
#pragma unroll
            for (int p = 0; p < 4; ++p) {
                int tp = 4 * p + j;  // i: j, f: 4+j, g: 8+j, o: 12+j
                int n = 16 * tp + (lane & 15);
                float4 wb = sWb[n];  // {wih0, wih1, bias, 0}
                v8f a;
#pragma unroll
                for (int r = 0; r < 8; ++r)
                    a[r] = wb.z + xm0[r] * wb.x + xm1[r] * wb.y;
                v16h b0 = load_tile16(sWhh, n, lane, 0);
                v16h b1 = load_tile16(sWhh, n, lane, 1);
                a = wmma_f16(hA0, b0, a);
                a = wmma_f16(hA1, b1, a);
                acc[p] = a;
            }
#pragma unroll
            for (int r = 0; r < 8; ++r) {
                float iv = sigf(acc[0][r]);
                float fv = sigf(acc[1][r]);
                float gv = tanh_fast(acc[2][r]);
                float ov = sigf(acc[3][r]);
                float c = fv * cs[j][r] + iv * gv;
                cs[j][r] = c;
                float h = ov * tanh_fast(c);
                int m = r + mhalf;
                int n = 16 * j + (lane & 15);
                sh[wave][m * 64 + n] = (_Float16)h;
            }
        }
    }

    // Final hidden state: coalesced LDS -> global copy (16 x 64 halves).
    {
        uint4* dst = (uint4*)(featH + (size_t)tile * 16 * 64);
        const uint4* src = (const uint4*)sh[wave];
        for (int i = lane; i < 128; i += 32) dst[i] = src[i];
    }
}

// ---------------------------------------------------------------------------
// Per-layer W -> fp16 transposed [out_col n][k], zero-padded rows.
// ---------------------------------------------------------------------------
__global__ void wprep_kernel(const float* __restrict__ W, _Float16* __restrict__ wT,
                             int FOUT) {
    int idx = blockIdx.x * blockDim.x + threadIdx.x;
    if (idx >= 64 * 64) return;
    int n = idx >> 6, k = idx & 63;
    wT[idx] = (n < FOUT) ? (_Float16)W[k * FOUT + n] : (_Float16)0.f;
}

// ---------------------------------------------------------------------------
// xs = feat(fp16) @ W(fp16), fp32 out. One wave per 16-node tile, WMMA.
// ---------------------------------------------------------------------------
template <int FOUT>
__global__ __launch_bounds__(32) void gemm_xs_kernel(
    const _Float16* __restrict__ featH,  // [N,64]
    const _Float16* __restrict__ wT16,   // [64(pad),64]
    float* __restrict__ xs,              // [N,FOUT]
    int N) {
    const int lane = threadIdx.x;
    const int base = blockIdx.x * 16;
    constexpr int NT = (FOUT + 15) / 16;
    const _Float16* atile = featH + (size_t)base * 64;
    v16h a0 = load_tile16(atile, lane & 15, lane, 0);
    v16h a1 = load_tile16(atile, lane & 15, lane, 1);
    const int mhalf = (lane >> 4) << 3;
#pragma unroll
    for (int tI = 0; tI < NT; ++tI) {
        int n = 16 * tI + (lane & 15);
        v16h b0 = load_tile16(wT16, n, lane, 0);
        v16h b1 = load_tile16(wT16, n, lane, 1);
        v8f acc;
#pragma unroll
        for (int r = 0; r < 8; ++r) acc[r] = 0.f;
        acc = wmma_f16(a0, b0, acc);
        acc = wmma_f16(a1, b1, acc);
        if (FOUT >= 16 || (lane & 15) < FOUT) {
#pragma unroll
            for (int r = 0; r < 8; ++r) {
                int m = r + mhalf;
                xs[(size_t)(base + m) * FOUT + n] = acc[r];
            }
        }
    }
}

// a_src[n] = xs[n,:].att_src ; a_dst[n] = xs[n,:].att_dst
template <int FOUT>
__global__ void attn_node_kernel(const float* __restrict__ xs,
                                 const float* __restrict__ attS,
                                 const float* __restrict__ attD,
                                 float* __restrict__ asrc, float* __restrict__ adst,
                                 int N) {
    int n = blockIdx.x * blockDim.x + threadIdx.x;
    if (n >= N) return;
    float s1 = 0.f, s2 = 0.f;
#pragma unroll
    for (int f = 0; f < FOUT; ++f) {
        float v = xs[(size_t)n * FOUT + f];
        s1 += v * attS[f];
        s2 += v * attD[f];
    }
    asrc[n] = s1;
    adst[n] = s2;
}

// we2 = W_e @ att_e  (edge_attr dim is 2)
__global__ void we2_prep_kernel(const float* __restrict__ We,
                                const float* __restrict__ atte,
                                float* __restrict__ we2, int F) {
    if (blockIdx.x == 0 && threadIdx.x == 0) {
        float w0 = 0.f, w1 = 0.f;
        for (int f = 0; f < F; ++f) {
            w0 += We[f] * atte[f];
            w1 += We[F + f] * atte[f];
        }
        we2[0] = w0;
        we2[1] = w1;
    }
}

template <int FOUT>
__global__ void layer_init_kernel(unsigned* __restrict__ amax, float* __restrict__ denom,
                                  float* __restrict__ out, const float* __restrict__ bias,
                                  float* __restrict__ stats, int N) {
    int n = blockIdx.x * blockDim.x + threadIdx.x;
    if (n >= N) return;
    amax[n] = ENC_NEG_INF;
    denom[n] = 0.f;
#pragma unroll
    for (int f = 0; f < FOUT; ++f) out[(size_t)n * FOUT + f] = bias[f];
    if (n == 0) { stats[0] = 0.f; stats[1] = 0.f; }
}

// Pass 1: alpha = leaky_relu(a_src[s]+a_dst[d]+ea.we2); segment max via atomicMax.
__global__ void edge_alpha_kernel(const int* __restrict__ ei,
                                  const float* __restrict__ eattr,
                                  const float* __restrict__ asrc,
                                  const float* __restrict__ adst,
                                  const float* __restrict__ we2,
                                  float* __restrict__ alpha,
                                  unsigned* __restrict__ amax, int E) {
    int e = blockIdx.x * blockDim.x + threadIdx.x;
    if (e >= E) return;
    int s = ei[e], d = ei[E + e];
    float a = asrc[s] + adst[d] + eattr[2 * e] * we2[0] + eattr[2 * e + 1] * we2[1];
    a = (a > 0.f) ? a : 0.2f * a;
    alpha[e] = a;
    atomicMax(amax + d, encf(a));
}

// Pass 2: ex = exp(alpha - amax[d]); denom[d] += ex  (alpha buffer now holds ex).
__global__ void edge_exp_kernel(const int* __restrict__ ei, float* __restrict__ alpha,
                                const unsigned* __restrict__ amax,
                                float* __restrict__ denom, int E) {
    int e = blockIdx.x * blockDim.x + threadIdx.x;
    if (e >= E) return;
    int d = ei[E + e];
    float ex = fexp(alpha[e] - decf(amax[d]));
    alpha[e] = ex;
    atomicAdd(denom + d, ex);
}

// Pass 3 (F=64): 16 lanes per edge, float4 gather + 4 float atomics each.
__global__ void edge_scatter64_kernel(const int* __restrict__ ei,
                                      const float* __restrict__ ex,
                                      const float* __restrict__ denom,
                                      const float* __restrict__ xs,
                                      float* __restrict__ out, int E) {
    int gid = blockIdx.x * blockDim.x + threadIdx.x;
    int e = gid >> 4;
    int q = gid & 15;
    if (e >= E) return;
    int s = ei[e], d = ei[E + e];
    __builtin_prefetch(xs + (size_t)s * 64, 0, 0);  // global_prefetch_b8
    float w = ex[e] * frcp(denom[d] + 1e-16f);
    float4 v = ((const float4*)(xs + (size_t)s * 64))[q];
    float* o = out + (size_t)d * 64 + q * 4;
    atomicAdd(o + 0, v.x * w);
    atomicAdd(o + 1, v.y * w);
    atomicAdd(o + 2, v.z * w);
    atomicAdd(o + 3, v.w * w);
}

// Pass 3 (F=2): one lane per edge.
__global__ void edge_scatter2_kernel(const int* __restrict__ ei,
                                     const float* __restrict__ ex,
                                     const float* __restrict__ denom,
                                     const float* __restrict__ xs,
                                     float* __restrict__ out, int E) {
    int e = blockIdx.x * blockDim.x + threadIdx.x;
    if (e >= E) return;
    int s = ei[e], d = ei[E + e];
    float w = ex[e] * frcp(denom[d] + 1e-16f);
    atomicAdd(out + 2 * d + 0, xs[2 * s + 0] * w);
    atomicAdd(out + 2 * d + 1, xs[2 * s + 1] * w);
}

// Graph layernorm: global sum/sumsq reduction, then apply + ReLU + fp16 copy.
__global__ void ln_reduce_kernel(const float* __restrict__ v, float* __restrict__ stats,
                                 int n) {
    __shared__ float s1[256], s2[256];
    int tid = threadIdx.x;
    float a = 0.f, b = 0.f;
    for (int i = blockIdx.x * blockDim.x + tid; i < n; i += gridDim.x * blockDim.x) {
        float t = v[i];
        a += t;
        b += t * t;
    }
    s1[tid] = a;
    s2[tid] = b;
    __syncthreads();
    for (int off = 128; off > 0; off >>= 1) {
        if (tid < off) {
            s1[tid] += s1[tid + off];
            s2[tid] += s2[tid + off];
        }
        __syncthreads();
    }
    if (tid == 0) {
        atomicAdd(&stats[0], s1[0]);
        atomicAdd(&stats[1], s2[0]);
    }
}

__global__ void ln_apply_kernel(float* __restrict__ v, _Float16* __restrict__ featH,
                                const float* __restrict__ stats,
                                const float* __restrict__ w, const float* __restrict__ b,
                                int n) {
    int i = blockIdx.x * blockDim.x + threadIdx.x;
    if (i >= n) return;
    float inv = 1.0f / (float)n;
    float mu = stats[0] * inv;
    float var = stats[1] * inv - mu * mu;
    float rs = __builtin_amdgcn_rsqf(var + 1e-5f);
    int f = i & 63;
    float y = w[f] * (v[i] - mu) * rs + b[f];
    y = fmaxf(y, 0.f);
    v[i] = y;
    featH[i] = (_Float16)y;
}

// ---------------------------------------------------------------------------
// Host-side pipeline
// ---------------------------------------------------------------------------
extern "C" void kernel_launch(void* const* d_in, const int* in_sizes, int n_in,
                              void* d_out, int out_size, void* d_ws, size_t ws_size,
                              hipStream_t stream) {
    const float* x     = (const float*)d_in[0];
    const int*   ei    = (const int*)d_in[1];
    const float* eattr = (const float*)d_in[2];
    const float* Wih   = (const float*)d_in[3];
    const float* Whh   = (const float*)d_in[4];
    const float* bih   = (const float*)d_in[5];
    const float* bhh   = (const float*)d_in[6];
    const float* Wl[3]   = {(const float*)d_in[7],  (const float*)d_in[13], (const float*)d_in[19]};
    const float* attS[3] = {(const float*)d_in[8],  (const float*)d_in[14], (const float*)d_in[20]};
    const float* attD[3] = {(const float*)d_in[9],  (const float*)d_in[15], (const float*)d_in[21]};
    const float* We[3]   = {(const float*)d_in[10], (const float*)d_in[16], (const float*)d_in[22]};
    const float* attE[3] = {(const float*)d_in[11], (const float*)d_in[17], (const float*)d_in[23]};
    const float* bias[3] = {(const float*)d_in[12], (const float*)d_in[18], (const float*)d_in[24]};
    const float* lnw[2]  = {(const float*)d_in[25], (const float*)d_in[27]};
    const float* lnb[2]  = {(const float*)d_in[26], (const float*)d_in[28]};

    const int N = in_sizes[0] / 48;  // x: [N,24,2]
    const int E = in_sizes[2] / 2;   // edge_attr: [E,2]

    // Workspace carve-out (256B aligned slices).
    char* p = (char*)d_ws;
    auto carve = [&](size_t bytes) -> void* {
        void* r = (void*)p;
        p += (bytes + 255) & ~(size_t)255;
        return r;
    };
    _Float16* whh16 = (_Float16*)carve((size_t)256 * 64 * 2);
    float4*   wbB   = (float4*)carve(256 * 16);
    _Float16* wT16  = (_Float16*)carve((size_t)64 * 64 * 2);
    _Float16* featH = (_Float16*)carve((size_t)N * 64 * 2);
    float*    xs    = (float*)carve((size_t)N * 64 * 4);
    float*    outF  = (float*)carve((size_t)N * 64 * 4);
    float*    asrc  = (float*)carve((size_t)N * 4);
    float*    adst  = (float*)carve((size_t)N * 4);
    float*    alpha = (float*)carve((size_t)E * 4);
    unsigned* amax  = (unsigned*)carve((size_t)N * 4);
    float*    denom = (float*)carve((size_t)N * 4);
    float*    we2   = (float*)carve(8);
    float*    stats = (float*)carve(8);

    const int nb = (N + 255) / 256;
    const int eb = (E + 255) / 256;
    const int nt = N / 16;                              // 3125 full tiles
    const int lstm_blocks = (nt + LSTM_WAVES - 1) / LSTM_WAVES;

    // ---- LSTM ----
    lstm_prep_kernel<<<(256 * 64 + 255) / 256, 256, 0, stream>>>(Whh, Wih, bih, bhh,
                                                                 whh16, wbB);
    lstm_kernel<<<lstm_blocks, 32 * LSTM_WAVES, 0, stream>>>(x, wbB, whh16, featH, N);

    // ---- GAT layers ----
    for (int li = 0; li < 3; ++li) {
        const int FOUT = (li == 2) ? 2 : 64;
        float* out = (li == 2) ? (float*)d_out : outF;

        wprep_kernel<<<16, 256, 0, stream>>>(Wl[li], wT16, FOUT);
        if (FOUT == 64) {
            gemm_xs_kernel<64><<<nt, 32, 0, stream>>>(featH, wT16, xs, N);
            attn_node_kernel<64><<<nb, 256, 0, stream>>>(xs, attS[li], attD[li], asrc, adst, N);
            layer_init_kernel<64><<<nb, 256, 0, stream>>>(amax, denom, out, bias[li], stats, N);
        } else {
            gemm_xs_kernel<2><<<nt, 32, 0, stream>>>(featH, wT16, xs, N);
            attn_node_kernel<2><<<nb, 256, 0, stream>>>(xs, attS[li], attD[li], asrc, adst, N);
            layer_init_kernel<2><<<nb, 256, 0, stream>>>(amax, denom, out, bias[li], stats, N);
        }
        we2_prep_kernel<<<1, 1, 0, stream>>>(We[li], attE[li], we2, FOUT);

        edge_alpha_kernel<<<eb, 256, 0, stream>>>(ei, eattr, asrc, adst, we2, alpha, amax, E);
        edge_exp_kernel<<<eb, 256, 0, stream>>>(ei, alpha, amax, denom, E);
        if (FOUT == 64) {
            edge_scatter64_kernel<<<((size_t)E * 16 + 255) / 256, 256, 0, stream>>>(
                ei, alpha, denom, xs, out, E);
        } else {
            edge_scatter2_kernel<<<eb, 256, 0, stream>>>(ei, alpha, denom, xs, out, E);
        }

        if (li < 2) {
            ln_reduce_kernel<<<1024, 256, 0, stream>>>(out, stats, N * 64);
            ln_apply_kernel<<<(N * 64 + 255) / 256, 256, 0, stream>>>(
                out, featH, stats, lnw[li], lnb[li], N * 64);
        }
    }
    (void)n_in; (void)out_size; (void)ws_size;
}
